// BaselineGRU_18614388260938
// MI455X (gfx1250) — matrix operand order repaired and checked by reading, not compiled, not run on hardware
//
#include <hip/hip_runtime.h>
#include <hip/hip_bf16.h>
#include <math.h>

// ---------------------------------------------------------------------------
// 2-layer GRU + decoder on MI455X (gfx1250), wave32, v_wmma_f32_16x16x32_f16.
//
//  1) pack f32 weights -> f16 WMMA B-fragment layout; pack x -> A-fragments.
//  2) gates_x0 = x @ w_ih_l0^T + b_ih_l0   (25.8 GFLOP WMMA GEMM).
//  3) persistent recurrence: 64 blocks x 1 wave; each block stages its 288 KB
//     of gate weights into LDS (320 KB/WG on CDNA5) ONCE, then runs 65
//     software-pipelined phases (layer0 step p + layer1 step p-1) with one
//     grid barrier per phase. h kept as f32 (exact update) + packed f16
//     A-fragments (zero-convert WMMA A loads).
//  4) decoder = packed-packed WMMA GEMM from final h2 fragments -> d_out.
// ---------------------------------------------------------------------------

typedef __attribute__((ext_vector_type(16))) _Float16 v16h;
typedef __attribute__((ext_vector_type(8)))  float    v8f;

#define NB_REC 64

__device__ __forceinline__ v8f wmma16(v16h a, v16h b, v8f c) {
  return __builtin_amdgcn_wmma_f32_16x16x32_f16(false, a, false, b, (short)0, c,
                                                false, false);
}

__device__ __forceinline__ float sigm(float x) {
  return 1.0f / (1.0f + __expf(-x));
}

// Monotonic-counter grid barrier (64 single-wave blocks, all co-resident).
__device__ __forceinline__ void grid_barrier(unsigned* __restrict__ cnt,
                                             unsigned target) {
  __threadfence();
  if (threadIdx.x == 0) {
    atomicAdd(cnt, 1u);
    while (__hip_atomic_load(cnt, __ATOMIC_RELAXED, __HIP_MEMORY_SCOPE_AGENT) <
           target) {
      __builtin_amdgcn_s_sleep(2);
    }
  }
  __builtin_amdgcn_wave_barrier();
  __threadfence();
}

// ---------------- packing ----------------------------------------------------

// W: [N][K] f32 row-major -> f16 B-fragments (B = W^T).
// B(32x16): lanes 0-15 (n=lane) K={0..7,16..23}; lanes 16-31 K offset +8.
__global__ __launch_bounds__(256) void pack_b_kernel(
    const float* __restrict__ W, _Float16* __restrict__ Wp, int N, int K) {
  int gid = blockIdx.x * blockDim.x + threadIdx.x;
  int KB = K >> 5;
  int total = (N >> 4) * KB * 32;
  if (gid >= total) return;
  int lane = gid & 31;
  int kb = (gid >> 5) % KB;
  int ntile = (gid >> 5) / KB;
  int nrow = (ntile << 4) + (lane & 15);
  int koff = (lane < 16) ? 0 : 8;
  const float* src = W + (size_t)nrow * K + (kb << 5) + koff;
  _Float16* dst = Wp + (size_t)gid * 16;
#pragma unroll
  for (int e = 0; e < 8; ++e) {
    dst[e]     = (_Float16)src[e];
    dst[e + 8] = (_Float16)src[e + 16];
  }
}

// X: [M][K] f32 row-major -> f16 A-fragments.
// A(16x32): lanes 0-15 (m=lane) K={0..7,16..23}; lanes 16-31 K offset +8.
__global__ __launch_bounds__(256) void pack_a_kernel(
    const float* __restrict__ X, _Float16* __restrict__ Xp, int M, int K) {
  int gid = blockIdx.x * blockDim.x + threadIdx.x;
  int KB = K >> 5;
  int total = (M >> 4) * KB * 32;
  if (gid >= total) return;
  int lane = gid & 31;
  int kb = (gid >> 5) % KB;
  int mtile = (gid >> 5) / KB;
  int m = (mtile << 4) + (lane & 15);
  int ko = (lane < 16) ? 0 : 8;
  const float* src = X + (size_t)m * K + (kb << 5) + ko;
  _Float16* dst = Xp + (size_t)gid * 16;
#pragma unroll
  for (int e = 0; e < 8; ++e) {
    dst[e]     = (_Float16)src[e];
    dst[e + 8] = (_Float16)src[e + 16];
  }
}

// ---------------- packed x packed GEMM --------------------------------------

// C[M][ldc] = Ap x Bp + bias; one wave per 16x16 output tile.
__global__ __launch_bounds__(256) void gemm_pp_kernel(
    const _Float16* __restrict__ Ap, const _Float16* __restrict__ Bp,
    const float* __restrict__ bias, float* __restrict__ C, int Mt, int Nt,
    int KB, int ldc) {
  int wave = (blockIdx.x * blockDim.x + threadIdx.x) >> 5;
  int lane = threadIdx.x & 31;
  if (wave >= Mt * Nt) return;
  int mt = wave / Nt, nt = wave - mt * Nt;
  const v16h* ap = (const v16h*)Ap + (size_t)mt * KB * 32 + lane;
  const v16h* bp = (const v16h*)Bp + (size_t)nt * KB * 32 + lane;
  v8f acc = {};
  for (int kb = 0; kb < KB; ++kb)
    acc = wmma16(ap[kb * 32], bp[kb * 32], acc);
  int nn = (nt << 4) + (lane & 15);
  float bv = bias[nn];
  int mbase = (mt << 4) + ((lane < 16) ? 0 : 8);
#pragma unroll
  for (int v = 0; v < 8; ++v)
    C[(size_t)(mbase + v) * ldc + nn] = acc[v] + bv;
}

// ---------------- persistent fused 2-layer recurrence -----------------------

__global__ __launch_bounds__(32) void gru_rec_kernel(
    const float* __restrict__ gx0, const _Float16* __restrict__ whh0p,
    const _Float16* __restrict__ wih1p, const _Float16* __restrict__ whh1p,
    const float* __restrict__ bhh0, const float* __restrict__ bih1,
    const float* __restrict__ bhh1,
    float* __restrict__ h1_32a, float* __restrict__ h1_32b,
    float* __restrict__ h2_32a, float* __restrict__ h2_32b,
    _Float16* __restrict__ h1_16a, _Float16* __restrict__ h1_16b,
    _Float16* __restrict__ h2_16a, _Float16* __restrict__ h2_16b,
    unsigned* __restrict__ cnt) {
  const int H = 1024, G3 = 3072, T = 64, KB = 32;
  const int lane = threadIdx.x;
  const int j = blockIdx.x;              // hidden column tile 0..63
  const int n = (j << 4) + (lane & 15);  // this lane's hidden column
  const int mb = (lane < 16) ? 0 : 8;    // C-fragment row base (batch rows)

  // Per-lane constants for scatter into packed f16 A-fragment layout:
  // column n -> kb_n = n>>5, kk = n&31; lane' = ((kk&8)?16:0)+m;
  // e = ((kk&16)?8:0) + (kk&7).
  const int kb_n = n >> 5;
  const int kk = n & 31;
  const int laneSel = (kk & 8) ? 16 : 0;
  const int e_n = ((kk & 16) ? 8 : 0) + (kk & 7);
  const int fragBase = (kb_n * 32 + laneSel) * 16 + e_n;  // + m*16

  // ---- stage this block's 288 KB of gate-weight B-fragments into LDS ----
  // [mg = mat*3+gate][kb][lane][16 halves]; mat 0=w_hh_l0, 1=w_ih_l1, 2=w_hh_l1
  __shared__ _Float16 wlds[9 * 32 * 32 * 16];  // 294912 B (<= 320 KB/WG)
  {
    const _Float16* gsrc[3] = {whh0p, wih1p, whh1p};
#pragma unroll
    for (int mg = 0; mg < 9; ++mg) {
      const v16h* s = (const v16h*)gsrc[mg / 3] +
                      ((size_t)((mg % 3) * 64 + j) * KB) * 32 + lane;
      v16h* d = (v16h*)wlds + (size_t)mg * KB * 32 + lane;
      for (int kb = 0; kb < KB; ++kb) d[kb * 32] = s[kb * 32];
    }
  }
  __builtin_amdgcn_wave_barrier();

  // ---- zero "t = -1" state (parity buffers index 1) ----
  for (int i = j * 32 + lane; i < 16 * H; i += NB_REC * 32) {
    h1_32b[i] = 0.0f;
    h2_32b[i] = 0.0f;
    h1_16b[i] = (_Float16)0.0f;
    h2_16b[i] = (_Float16)0.0f;
  }
  unsigned bt = 0;
  grid_barrier(cnt, bt += NB_REC);

  float*    h1_32[2] = {h1_32a, h1_32b};  // index t&1 holds h?[t]
  float*    h2_32[2] = {h2_32a, h2_32b};
  _Float16* h1_16[2] = {h1_16a, h1_16b};
  _Float16* h2_16[2] = {h2_16a, h2_16b};

  const float br0 = bhh0[n], bz0 = bhh0[H + n], bn0 = bhh0[2 * H + n];
  const float bxr = bih1[n], bxz = bih1[H + n], bxn = bih1[2 * H + n];
  const float br1 = bhh1[n], bz1 = bhh1[H + n], bn1 = bhh1[2 * H + n];

  const v16h* wl = (const v16h*)wlds;

  // Software-pipelined phases: phase p runs layer0 step p and layer1 step p-1.
  for (int p = 0; p <= T; ++p) {
    if (p < T) {  // ---- layer 0, step t=p: gh = h1[p-1] @ w_hh_l0^T
      const int t = p;
      const v16h* hA = (const v16h*)h1_16[(t + 1) & 1] + lane;
      const float* hOld = h1_32[(t + 1) & 1];
      v8f ar = {}, az = {}, ag = {};
      for (int kb = 0; kb < KB; ++kb) {
        v16h a = hA[kb * 32];
        ar = wmma16(a, wl[(0 * KB + kb) * 32 + lane], ar);
        az = wmma16(a, wl[(1 * KB + kb) * 32 + lane], az);
        ag = wmma16(a, wl[(2 * KB + kb) * 32 + lane], ag);
      }
      float*    o32 = h1_32[t & 1];
      _Float16* o16 = h1_16[t & 1];
#pragma unroll
      for (int v = 0; v < 8; ++v) {
        int m = mb + v;
        const float* gx = gx0 + ((size_t)m * T + t) * G3;  // b_ih folded in
        float r  = sigm(gx[n] + ar[v] + br0);
        float z  = sigm(gx[H + n] + az[v] + bz0);
        float nn = tanhf(gx[2 * H + n] + r * (ag[v] + bn0));
        float hv = (1.0f - z) * nn + z * hOld[m * H + n];
        o32[m * H + n] = hv;
        o16[fragBase + m * 16] = (_Float16)hv;
      }
    }
    if (p > 0) {  // ---- layer 1, step t=p-1 (h1[t], h2[t-1] synced already)
      const int t = p - 1;
      const v16h* a1p = (const v16h*)h1_16[t & 1] + lane;
      const v16h* a2p = (const v16h*)h2_16[(t + 1) & 1] + lane;
      const float* hOld = h2_32[(t + 1) & 1];
      v8f xr = {}, xz = {}, xg = {}, hr = {}, hz = {}, hg = {};
      for (int kb = 0; kb < KB; ++kb) {
        v16h a1 = a1p[kb * 32];
        v16h a2 = a2p[kb * 32];
        xr = wmma16(a1, wl[(3 * KB + kb) * 32 + lane], xr);
        xz = wmma16(a1, wl[(4 * KB + kb) * 32 + lane], xz);
        xg = wmma16(a1, wl[(5 * KB + kb) * 32 + lane], xg);
        hr = wmma16(a2, wl[(6 * KB + kb) * 32 + lane], hr);
        hz = wmma16(a2, wl[(7 * KB + kb) * 32 + lane], hz);
        hg = wmma16(a2, wl[(8 * KB + kb) * 32 + lane], hg);
      }
      float*    o32 = h2_32[t & 1];
      _Float16* o16 = h2_16[t & 1];
#pragma unroll
      for (int v = 0; v < 8; ++v) {
        int m = mb + v;
        float r  = sigm(xr[v] + bxr + hr[v] + br1);
        float z  = sigm(xz[v] + bxz + hz[v] + bz1);
        float nn = tanhf(xg[v] + bxn + r * (hg[v] + bn1));
        float hv = (1.0f - z) * nn + z * hOld[m * H + n];
        o32[m * H + n] = hv;
        o16[fragBase + m * 16] = (_Float16)hv;
      }
    }
    grid_barrier(cnt, bt += NB_REC);
  }
  // final h2 is step t=63 -> parity 1 -> h2_16b / h2_32b.
}

__global__ void zero_cnt_kernel(unsigned* c) {
  if (threadIdx.x == 0) *c = 0;
}

// ---------------- host launch ------------------------------------------------

extern "C" void kernel_launch(void* const* d_in, const int* in_sizes, int n_in,
                              void* d_out, int out_size, void* d_ws,
                              size_t ws_size, hipStream_t stream) {
  (void)in_sizes; (void)n_in; (void)out_size; (void)ws_size;
  const int B = 16, T = 64, NN = 4096, H = 1024, G3 = 3 * H, M = B * T;

  const float* x    = (const float*)d_in[0];
  const float* wih0 = (const float*)d_in[1];
  const float* whh0 = (const float*)d_in[2];
  const float* bih0 = (const float*)d_in[3];
  const float* bhh0 = (const float*)d_in[4];
  const float* wih1 = (const float*)d_in[5];
  const float* whh1 = (const float*)d_in[6];
  const float* bih1 = (const float*)d_in[7];
  const float* bhh1 = (const float*)d_in[8];
  const float* decw = (const float*)d_in[9];
  const float* decb = (const float*)d_in[10];
  float* out = (float*)d_out;

  char* ws = (char*)d_ws;
  size_t off = 0;
  auto alloc = [&](size_t bytes) -> void* {
    void* p = ws + off;
    off = (off + bytes + 255) & ~(size_t)255;
    return p;
  };
  _Float16* wih0p = (_Float16*)alloc((size_t)G3 * NN * 2);  // 24 MB
  _Float16* whh0p = (_Float16*)alloc((size_t)G3 * H * 2);   //  6 MB
  _Float16* wih1p = (_Float16*)alloc((size_t)G3 * H * 2);   //  6 MB
  _Float16* whh1p = (_Float16*)alloc((size_t)G3 * H * 2);   //  6 MB
  _Float16* decp  = (_Float16*)alloc((size_t)NN * H * 2);   //  8 MB
  _Float16* xap   = (_Float16*)alloc((size_t)M * NN * 2);   //  8 MB
  float* gx0 = (float*)alloc((size_t)M * G3 * 4);           // 12 MB
  float* h1_32a = (float*)alloc((size_t)B * H * 4);
  float* h1_32b = (float*)alloc((size_t)B * H * 4);
  float* h2_32a = (float*)alloc((size_t)B * H * 4);
  float* h2_32b = (float*)alloc((size_t)B * H * 4);
  _Float16* h1_16a = (_Float16*)alloc((size_t)B * H * 2);
  _Float16* h1_16b = (_Float16*)alloc((size_t)B * H * 2);
  _Float16* h2_16a = (_Float16*)alloc((size_t)B * H * 2);
  _Float16* h2_16b = (_Float16*)alloc((size_t)B * H * 2);
  unsigned* cnt = (unsigned*)alloc(256);

  // 1) pack weights / input into WMMA fragment layouts (f32 -> f16)
  pack_b_kernel<<<(G3 / 16) * (NN / 32) * 32 / 256, 256, 0, stream>>>(
      wih0, wih0p, G3, NN);
  pack_b_kernel<<<(G3 / 16) * (H / 32) * 32 / 256, 256, 0, stream>>>(
      whh0, whh0p, G3, H);
  pack_b_kernel<<<(G3 / 16) * (H / 32) * 32 / 256, 256, 0, stream>>>(
      wih1, wih1p, G3, H);
  pack_b_kernel<<<(G3 / 16) * (H / 32) * 32 / 256, 256, 0, stream>>>(
      whh1, whh1p, G3, H);
  pack_b_kernel<<<(NN / 16) * (H / 32) * 32 / 256, 256, 0, stream>>>(
      decw, decp, NN, H);
  pack_a_kernel<<<(M / 16) * (NN / 32) * 32 / 256, 256, 0, stream>>>(
      x, xap, M, NN);

  // 2) gates_x0 = x @ w_ih_l0^T + b_ih_l0   [1024 x 3072]
  gemm_pp_kernel<<<(M / 16) * (G3 / 16) * 32 / 256, 256, 0, stream>>>(
      xap, wih0p, bih0, gx0, M / 16, G3 / 16, NN / 32, G3);

  // 3) fused persistent 2-layer recurrence (weights LDS-resident per block)
  zero_cnt_kernel<<<1, 32, 0, stream>>>(cnt);
  gru_rec_kernel<<<NB_REC, 32, 0, stream>>>(
      gx0, whh0p, wih1p, whh1p, bhh0, bih1, bhh1, h1_32a, h1_32b, h2_32a,
      h2_32b, h1_16a, h1_16b, h2_16a, h2_16b, cnt);

  // 4) decoder: out[16][4096] = h2[63] @ dec_w^T + dec_b (packed A = h2_16b)
  gemm_pp_kernel<<<(NN / 16) * 32 / 256, 256, 0, stream>>>(
      h2_16b, decp, decb, out, 1, NN / 16, H / 32, NN);
}
